// DPLSTM_37915971289788
// MI455X (gfx1250) — matrix operand order repaired and checked by reading, not compiled, not run on hardware
//
#include <hip/hip_runtime.h>

#define TSEQ   512
#define BATCH  128
#define HID    256
#define NLAYER 2
#define NWG    16     // one WG per 16-column chunk of H (also fits one 16-WG cluster)
#define NTHR   256    // 8 wave32 -> one 16-row batch tile per wave

typedef __attribute__((ext_vector_type(16))) __bf16 v16bf;
typedef __attribute__((ext_vector_type(8)))  float  v8f;

union FragBF { v16bf v; unsigned long long d[4]; };

__device__ __forceinline__ unsigned short f2bf(float f) {
  unsigned u = __float_as_uint(f);
  u += 0x7FFFu + ((u >> 16) & 1u);          // round-to-nearest-even
  return (unsigned short)(u >> 16);
}
__device__ __forceinline__ float sigm(float x) { return 1.0f / (1.0f + __expf(-x)); }
__device__ __forceinline__ float tanh_(float x) {
  float e = __expf(-2.0f * x);
  return (1.0f - e) / (1.0f + e);
}

// A fragment: 16x32 bf16 tile (rows row0..row0+15, k-slice k0..k0+31), LDS row-major stride HID.
// Layout (ISA 7.12.2, 16-bit A 16x32): lane half 0 -> K {k0+0..7, k0+16..23}, half 1 -> +8.
__device__ __forceinline__ FragBF loadA(const unsigned short* lds, int row0, int k0, int r, int hf) {
  FragBF f;
  const unsigned short* p = lds + (row0 + r) * HID + k0 + hf * 8;
  f.d[0] = *(const unsigned long long*)(p +  0);
  f.d[1] = *(const unsigned long long*)(p +  4);
  f.d[2] = *(const unsigned long long*)(p + 16);
  f.d[3] = *(const unsigned long long*)(p + 20);
  return f;
}
// B fragment: 32x16 (K x N) tile of W^T; LDS holds W rows (one gate-column per row) x k.
// Layout: lane n = lane%16, half h holds K = k0 + h*8 + {0..7, 16..23} (matches A).
__device__ __forceinline__ FragBF loadB(const unsigned short* lds, int row0, int k0, int n, int hf) {
  FragBF f;
  const unsigned short* p = lds + (row0 + n) * HID + k0 + hf * 8;
  f.d[0] = *(const unsigned long long*)(p +  0);
  f.d[1] = *(const unsigned long long*)(p +  4);
  f.d[2] = *(const unsigned long long*)(p + 16);
  f.d[3] = *(const unsigned long long*)(p + 20);
  return f;
}

// One LSTM layer-step for this wave's 16x16 tile of h.
__device__ __forceinline__ void lstm_phase(
    const unsigned short* __restrict__ inpL,   // LDS [128][256] bf16 (layer input)
    const unsigned short* __restrict__ hstL,   // LDS [128][256] bf16 (h state)
    const unsigned short* __restrict__ wihL,   // LDS [64][256] bf16 (4 gates x 16 cols)
    const unsigned short* __restrict__ whhL,   // LDS [64][256] bf16
    const float* __restrict__ biasL,           // LDS [4][16] (b_ih + b_hh)
    float* creg, float* hout, int wave, int r, int hf) {
  v8f acc[4];
#pragma unroll
  for (int g = 0; g < 4; ++g)
#pragma unroll
    for (int j = 0; j < 8; ++j) acc[g][j] = 0.0f;

  const int m0 = wave * 16;
  for (int k0 = 0; k0 < HID; k0 += 32) {
    FragBF ax = loadA(inpL, m0, k0, r, hf);
    FragBF ah = loadA(hstL, m0, k0, r, hf);
#pragma unroll
    for (int g = 0; g < 4; ++g) {
      FragBF bi = loadB(wihL, g * 16, k0, r, hf);
      acc[g] = __builtin_amdgcn_wmma_f32_16x16x32_bf16(false, ax.v, false, bi.v,
                                                       (short)0, acc[g], false, false);
      FragBF bh = loadB(whhL, g * 16, k0, r, hf);
      acc[g] = __builtin_amdgcn_wmma_f32_16x16x32_bf16(false, ah.v, false, bh.v,
                                                       (short)0, acc[g], false, false);
    }
  }
  const float bi_ = biasL[0 * 16 + r];
  const float bf_ = biasL[1 * 16 + r];
  const float bg_ = biasL[2 * 16 + r];
  const float bo_ = biasL[3 * 16 + r];
#pragma unroll
  for (int j = 0; j < 8; ++j) {
    float gi = sigm (acc[0][j] + bi_);
    float gf = sigm (acc[1][j] + bf_);
    float gg = tanh_(acc[2][j] + bg_);
    float go = sigm (acc[3][j] + bo_);
    float c  = gf * creg[j] + gi * gg;
    creg[j]  = c;
    hout[j]  = go * tanh_(c);
  }
}

// Grid-wide barrier between layer-steps.
//  - Cluster fast path: hardware cluster barrier (-3). One wave per WG signals,
//    all waves wait (ISA §6.6). Only valid when the dispatch runs as a cluster.
//  - Fallback: atomic counter + generation spin through L2.
__device__ __forceinline__ void grid_barrier(unsigned* cnt, unsigned* gen, int incluster) {
  __threadfence();          // make this thread's global stores device-visible
  __syncthreads();
  if (incluster) {
    if (threadIdx.x < 32) {
      asm volatile("s_barrier_signal -3" ::: "memory");
    }
    asm volatile("s_barrier_wait -3" ::: "memory");
  } else if (threadIdx.x == 0) {
    volatile unsigned* vg = gen;
    unsigned old  = *vg;
    unsigned prev = atomicAdd(cnt, 1u);
    if (prev == NWG - 1) {
      *cnt = 0u;
      __threadfence();
      atomicAdd(gen, 1u);
    } else {
      while (*vg == old) { __builtin_amdgcn_s_sleep(1); }
    }
    __threadfence();
  }
  __syncthreads();
}

// fp32 global -> bf16 LDS
__device__ __forceinline__ void stage_x(unsigned short* dst, const float* __restrict__ src, int tid) {
  for (int i = tid * 4; i < BATCH * HID; i += NTHR * 4) {
    float4 v = *(const float4*)(src + i);
    unsigned long long pk = (unsigned long long)f2bf(v.x)
                          | ((unsigned long long)f2bf(v.y) << 16)
                          | ((unsigned long long)f2bf(v.z) << 32)
                          | ((unsigned long long)f2bf(v.w) << 48);
    *(unsigned long long*)(dst + i) = pk;
  }
}
// bf16 global -> bf16 LDS
__device__ __forceinline__ void stage_cp(unsigned short* dst, const unsigned short* __restrict__ src, int tid) {
  for (int i = tid * 8; i < BATCH * HID; i += NTHR * 8) {
    uint4 v = *(const uint4*)(src + i);
    *(uint4*)(dst + i) = v;
  }
}

__global__ void __launch_bounds__(NTHR, 1) lstm_persistent(
    const float* __restrict__ x,   const float* __restrict__ Wih,
    const float* __restrict__ Whh, const float* __restrict__ bih,
    const float* __restrict__ bhh, float* __restrict__ out,
    const unsigned short* __restrict__ xbf,  // optional pre-converted bf16 x (may be null)
    unsigned short* __restrict__ h0buf,      // [2][128*256] bf16 double-buffered
    unsigned short* __restrict__ h1buf,      // [2][128*256]
    unsigned* barcnt, unsigned* bargen) {
  extern __shared__ unsigned char smem[];
  unsigned short* wihL = (unsigned short*)smem;             // [2][64][256]
  unsigned short* whhL = wihL + NLAYER * 64 * HID;          // [2][64][256]
  unsigned short* inpL = whhL + NLAYER * 64 * HID;          // [128][256]
  unsigned short* hstL = inpL + BATCH * HID;                // [128][256]
  float*          biasL = (float*)(hstL + BATCH * HID);     // [2][4][16]

  const int wg  = blockIdx.x;
  const int tid = threadIdx.x;

  int incluster = 0;
#if __has_builtin(__builtin_amdgcn_cluster_id_x)
  incluster = (__builtin_amdgcn_cluster_id_x() != 0);       // uniform across the (single) cluster
#endif

  // ---- one-time: stage this WG's weight slices (bf16) + fused biases into LDS ----
  for (int idx = tid; idx < NLAYER * 64 * HID; idx += NTHR) {
    int l    = idx / (64 * HID);
    int rem  = idx - l * (64 * HID);
    int q    = rem / HID;                 // g*16 + jj
    int k    = rem - q * HID;
    int g    = q >> 4, jj = q & 15;
    int grow = g * HID + wg * 16 + jj;    // gate row in [0,1024)
    wihL[idx] = f2bf(Wih[((size_t)l * 4 * HID + grow) * HID + k]);
    whhL[idx] = f2bf(Whh[((size_t)l * 4 * HID + grow) * HID + k]);
  }
  if (tid < NLAYER * 64) {
    int l = tid >> 6, q = tid & 63;
    int g = q >> 4, jj = q & 15;
    int grow = g * HID + wg * 16 + jj;
    biasL[tid] = bih[l * 4 * HID + grow] + bhh[l * 4 * HID + grow];
  }
  __syncthreads();

  const int wave = tid >> 5, lane = tid & 31;
  const int r = lane & 15, hf = lane >> 4;
  const int col = wg * 16 + r;
  const size_t OUT_HN = (size_t)TSEQ * BATCH * HID;
  const size_t OUT_CN = OUT_HN + (size_t)NLAYER * BATCH * HID;

  float c0[8], c1[8], h8[8];
#pragma unroll
  for (int j = 0; j < 8; ++j) { c0[j] = 0.0f; c1[j] = 0.0f; }

  for (int t = 0; t < TSEQ; ++t) {
    const int p = t & 1;

    // ---------------- layer 0 ----------------
    if (xbf) stage_cp(inpL, xbf + (size_t)t * BATCH * HID, tid);
    else     stage_x (inpL, x   + (size_t)t * BATCH * HID, tid);
    stage_cp(hstL, h0buf + p * BATCH * HID, tid);
    __syncthreads();
    lstm_phase(inpL, hstL, wihL, whhL, biasL, c0, h8, wave, r, hf);
    {
      unsigned short* hb = h0buf + (p ^ 1) * BATCH * HID;
#pragma unroll
      for (int j = 0; j < 8; ++j) {
        int M = wave * 16 + hf * 8 + j;
        hb[M * HID + col] = f2bf(h8[j]);
      }
      if (t == TSEQ - 1) {
#pragma unroll
        for (int j = 0; j < 8; ++j) {
          int M = wave * 16 + hf * 8 + j;
          out[OUT_HN + (size_t)M * HID + col] = h8[j];
          out[OUT_CN + (size_t)M * HID + col] = c0[j];
        }
      }
    }
    grid_barrier(barcnt, bargen, incluster);

    // ---------------- layer 1 ----------------
    stage_cp(inpL, h0buf + (p ^ 1) * BATCH * HID, tid);
    stage_cp(hstL, h1buf + p * BATCH * HID, tid);
    if (t + 1 < TSEQ) {  // pull next x_t toward L2 while we compute
      if (xbf) __builtin_prefetch(xbf + (size_t)(t + 1) * BATCH * HID + tid * 128, 0, 1);
      else     __builtin_prefetch(x   + (size_t)(t + 1) * BATCH * HID + tid * 128, 0, 1);
    }
    __syncthreads();
    lstm_phase(inpL, hstL, wihL + 64 * HID, whhL + 64 * HID, biasL + 64, c1, h8, wave, r, hf);
    {
      unsigned short* hb = h1buf + (p ^ 1) * BATCH * HID;
#pragma unroll
      for (int j = 0; j < 8; ++j) {
        int M = wave * 16 + hf * 8 + j;
        float h = h8[j];
        hb[M * HID + col] = f2bf(h);
        out[(size_t)t * BATCH * HID + (size_t)M * HID + col] = h;  // sequence output
      }
      if (t == TSEQ - 1) {
#pragma unroll
        for (int j = 0; j < 8; ++j) {
          int M = wave * 16 + hf * 8 + j;
          out[OUT_HN + (size_t)BATCH * HID + (size_t)M * HID + col] = h8[j];
          out[OUT_CN + (size_t)BATCH * HID + (size_t)M * HID + col] = c1[j];
        }
      }
    }
    grid_barrier(barcnt, bargen, incluster);
  }
}

// zero barrier state + h double-buffers
__global__ void init_ws(unsigned* ws, int n) {
  int i = blockIdx.x * blockDim.x + threadIdx.x;
  if (i < n) ws[i] = 0u;
}

// one-shot fp32 -> bf16 conversion of the whole input sequence
__global__ void xcvt(const float* __restrict__ x, unsigned short* __restrict__ xbf, int n4) {
  int i = blockIdx.x * blockDim.x + threadIdx.x;
  if (i < n4) {
    float4 v = *(const float4*)(x + (size_t)i * 4);
    unsigned long long pk = (unsigned long long)f2bf(v.x)
                          | ((unsigned long long)f2bf(v.y) << 16)
                          | ((unsigned long long)f2bf(v.z) << 32)
                          | ((unsigned long long)f2bf(v.w) << 48);
    *(unsigned long long*)(xbf + (size_t)i * 4) = pk;
  }
}

extern "C" void kernel_launch(void* const* d_in, const int* in_sizes, int n_in,
                              void* d_out, int out_size, void* d_ws, size_t ws_size,
                              hipStream_t stream) {
  const float* x   = (const float*)d_in[0];
  const float* Wih = (const float*)d_in[1];
  const float* Whh = (const float*)d_in[2];
  const float* bih = (const float*)d_in[3];
  const float* bhh = (const float*)d_in[4];
  float* out = (float*)d_out;

  unsigned*       barcnt = (unsigned*)d_ws;
  unsigned*       bargen = barcnt + 1;
  unsigned short* h0buf  = (unsigned short*)((char*)d_ws + 256);
  unsigned short* h1buf  = h0buf + 2 * BATCH * HID;

  const size_t baseBytes = 256 + (size_t)2 * 2 * BATCH * HID * sizeof(unsigned short); // 262,400
  const size_t xbfBytes  = (size_t)TSEQ * BATCH * HID * sizeof(unsigned short);        // 32 MB
  unsigned short* xbf = nullptr;
  if (ws_size >= baseBytes + xbfBytes) {
    xbf = (unsigned short*)((char*)d_ws + baseBytes);   // baseBytes is 256-aligned
    const int n4 = TSEQ * BATCH * HID / 4;
    xcvt<<<dim3((n4 + 255) / 256), dim3(256), 0, stream>>>(x, xbf, n4);
  }

  const int initN = (int)(baseBytes / 4);
  init_ws<<<dim3((initN + 255) / 256), dim3(256), 0, stream>>>((unsigned*)d_ws, initN);

  // LDS: 2x W_ih slice + 2x W_hh slice (bf16) + inp + hstate staging + biases
  const size_t smemBytes =
      (size_t)(NLAYER * 64 * HID + NLAYER * 64 * HID + BATCH * HID + BATCH * HID) * 2
      + (size_t)NLAYER * 4 * 16 * 4;   // 262,656 B of the 320 KB/WGP

  lstm_persistent<<<dim3(NWG), dim3(NTHR), smemBytes, stream>>>(
      x, Wih, Whh, bih, bhh, out, xbf, h0buf, h1buf, barcnt, bargen);
}
